// SelfAttentionWithLeads_30820685316215
// MI455X (gfx1250) — matrix operand order repaired and checked by reading, not compiled, not run on hardware
//
#include <hip/hip_runtime.h>
#include <hip/hip_bf16.h>

// ---------------------------------------------------------------------------
// CDNA5 (gfx1250) fused "self-attention with leads".
// bf16 WMMA (v_wmma_f32_16x16x32_bf16), wave32, 320KB-class LDS usage.
// ---------------------------------------------------------------------------

typedef __attribute__((ext_vector_type(16))) __bf16 v16bf;
typedef __attribute__((ext_vector_type(8)))  float  v8f;

struct alignas(16) U4 { unsigned int x[4]; };

__device__ __forceinline__ v8f zero8() {
    v8f z = {0.f, 0.f, 0.f, 0.f, 0.f, 0.f, 0.f, 0.f};
    return z;
}

__device__ __forceinline__ v8f wmma_bf16(v16bf a, v16bf b, v8f c) {
    return __builtin_amdgcn_wmma_f32_16x16x32_bf16(
        /*neg_a=*/false, a, /*neg_b=*/false, b,
        /*c_mod=*/(short)0, c, /*reuse_a=*/false, /*reuse_b=*/false);
}

// A-matrix 16x32 bf16 fragment from f32 row-major source.
// lane m = L%16; K-half = L/16; VGPR v -> K = 16*half + 2v + {0,1}
// (contiguous order, matching the B loaders); K pairs contiguous.
__device__ __forceinline__ v16bf load_a_from_f32(const float* base, int rowStride, int lane) {
    const int m  = lane & 15;
    const int kh = lane >> 4;
    const float* r = base + m * rowStride + kh * 16;
    v16bf a;
#pragma unroll
    for (int v = 0; v < 8; ++v) {
        const int kb = 2 * v;
        float2 p = *(const float2*)(r + kb);
        a[2 * v]     = (__bf16)p.x;
        a[2 * v + 1] = (__bf16)p.y;
    }
    return a;
}

// A-matrix 16x32 bf16 fragment from bf16 row-major source (global or LDS).
__device__ __forceinline__ v16bf load_a_from_bf16(const __bf16* base, int rowStride, int lane) {
    const int m  = lane & 15;
    const int kh = lane >> 4;
    const __bf16* r = base + m * rowStride + kh * 16;
    union { unsigned int u[8]; v16bf v; } cv;
#pragma unroll
    for (int t = 0; t < 8; ++t) {
        const int kb = 2 * t;
        cv.u[t] = *(const unsigned int*)(r + kb);  // packs K=kb,kb+1
    }
    return cv.v;
}

// B-matrix 32x16 bf16 fragment from an "n-major" source S[n][k] (each output
// column's K values contiguous). contiguous order used here: lane n = L%16,
// K-half = L/16; VGPR v -> K = 16*half + 2v + {0,1}. -> one b128 pair.
__device__ __forceinline__ v16bf load_b_nk(const __bf16* base, int rowStride, int lane) {
    const int n  = lane & 15;
    const int kh = lane >> 4;
    const __bf16* r = base + n * rowStride + kh * 16;
    union { U4 q[2]; v16bf v; } cv;
    cv.q[0] = *(const U4*)(r);
    cv.q[1] = *(const U4*)(r + 8);
    return cv.v;
}

// B-matrix fragment from k-major f32 weights W[k][ncols] (strided gather + cvt).
__device__ __forceinline__ v16bf load_b_weights(const float* W, int ncols, int nbase,
                                                int k0, int lane) {
    const int n  = nbase + (lane & 15);
    const int kh = lane >> 4;
    v16bf b;
#pragma unroll
    for (int v = 0; v < 8; ++v) {
        const int k = k0 + 16 * kh + 2 * v;
        b[2 * v]     = (__bf16)W[(size_t)k * ncols + n];
        b[2 * v + 1] = (__bf16)W[(size_t)(k + 1) * ncols + n];
    }
    return b;
}

// ---------------------------------------------------------------------------
// Kernel 1: fused projection GEMM  [pixels x 256] * [256 x (32|32|256)]
// writes f_fl[b][n][96], g_fl[b][n][96] and transposed h_t[b][c'][1024], bf16.
// Grid: B*12*2 = 768 blocks, 128 threads (4 waves). Each block: 128 pixels.
// ---------------------------------------------------------------------------
__global__ void __launch_bounds__(128)
salw_proj_kernel(const float* __restrict__ x,
                 const float* __restrict__ Wf,
                 const float* __restrict__ Wg,
                 const float* __restrict__ Wh,
                 __bf16* __restrict__ f_fl,
                 __bf16* __restrict__ g_fl,
                 __bf16* __restrict__ h_t) {
    const int blk   = blockIdx.x;
    const int b     = blk / 24;
    const int rem   = blk % 24;
    const int hidx  = rem >> 1;       // 0..11
    const int wBase = (rem & 1) * 128;
    const int sp    = hidx >> 2;      // lead group 0..2
    const int hh    = hidx & 3;       // row within group 0..3

    const int lane = threadIdx.x & 31;
    const int wv   = threadIdx.x >> 5;  // 0..3

#pragma unroll
    for (int ctL = 0; ctL < 5; ++ctL) {
        const int g = wv * 5 + ctL;     // global col tile 0..19 of 320 outputs
        const float* Wm;
        int ncols, nbase, dest;
        if (g < 2)      { Wm = Wf; ncols = 32;  nbase = g * 16;        dest = 0; }
        else if (g < 4) { Wm = Wg; ncols = 32;  nbase = (g - 2) * 16;  dest = 1; }
        else            { Wm = Wh; ncols = 256; nbase = (g - 4) * 16;  dest = 2; }

        // hoist the 8 K-step weight fragments for this column tile
        v16bf bw[8];
#pragma unroll
        for (int kt = 0; kt < 8; ++kt)
            bw[kt] = load_b_weights(Wm, ncols, nbase, kt * 32, lane);

        for (int mt = 0; mt < 8; ++mt) {
            const float* xb = x + ((size_t)((b * 12 + hidx) * 256 + wBase + mt * 16)) * 256;
            v8f acc = zero8();
#pragma unroll
            for (int kt = 0; kt < 8; ++kt) {
                v16bf a = load_a_from_f32(xb + kt * 32, 256, lane);
                acc = wmma_bf16(a, bw[kt], acc);
            }
            // scatter D fragment into the flattened / transposed layouts
            const int col = nbase + (lane & 15);
            const int kh  = lane >> 4;
#pragma unroll
            for (int v = 0; v < 8; ++v) {
                const int wg = wBase + mt * 16 + v + 8 * kh;   // pixel w
                const int n  = hh * 256 + wg;                  // flattened index
                const __bf16 val = (__bf16)acc[v];
                if (dest == 0) {
                    f_fl[(size_t)(b * 1024 + n) * 96 + sp * 32 + col] = val;
                } else if (dest == 1) {
                    g_fl[(size_t)(b * 1024 + n) * 96 + sp * 32 + col] = val;
                } else {
                    h_t[(size_t)(b * 768 + sp * 256 + col) * 1024 + n] = val;
                }
            }
        }
    }
}

// ---------------------------------------------------------------------------
// Kernel 2: fused  s = g.f^T  -> softmax -> o = beta.h  -> gamma*o + x
// Grid: B * (1024/32) = 1024 blocks, 256 threads (8 waves).
// LDS: 32x1024 f32 scores (128KB) + 32x1024 bf16 beta (64KB) = 192KB (CDNA5).
// ---------------------------------------------------------------------------
__global__ void __launch_bounds__(256)
salw_attn_kernel(const float* __restrict__ x,
                 const __bf16* __restrict__ f_fl,
                 const __bf16* __restrict__ g_fl,
                 const __bf16* __restrict__ h_t,
                 const float* __restrict__ gammaP,
                 float* __restrict__ out) {
    extern __shared__ char smem[];
    float*  sS = (float*)smem;                        // [32][1024]
    __bf16* sB = (__bf16*)(smem + 32 * 1024 * 4);     // [32][1024]

    const int b     = blockIdx.x >> 5;
    const int nBase = (blockIdx.x & 31) * 32;
    const int lane  = threadIdx.x & 31;
    const int wv    = threadIdx.x >> 5;   // 0..7

    // ---------------- stage 2: s = g_fl . f_fl^T (K = 96) ----------------
    v16bf aG[2][3];
    {
        const __bf16* gb = g_fl + (size_t)(b * 1024 + nBase) * 96;
#pragma unroll
        for (int mt = 0; mt < 2; ++mt)
#pragma unroll
            for (int kt = 0; kt < 3; ++kt)
                aG[mt][kt] = load_a_from_bf16(gb + mt * 16 * 96 + kt * 32, 96, lane);
    }

#pragma unroll
    for (int ctL = 0; ctL < 8; ++ctL) {
        const int colBase = (wv * 8 + ctL) * 16;      // 0..1008
        const __bf16* fb = f_fl + (size_t)(b * 1024 + colBase) * 96;
        v8f acc0 = zero8(), acc1 = zero8();
#pragma unroll
        for (int kt = 0; kt < 3; ++kt) {
            v16bf bF = load_b_nk(fb + kt * 32, 96, lane);
            acc0 = wmma_bf16(aG[0][kt], bF, acc0);
            acc1 = wmma_bf16(aG[1][kt], bF, acc1);
        }
        const int nCol = colBase + (lane & 15);
        const int kh   = lane >> 4;
#pragma unroll
        for (int v = 0; v < 8; ++v) {
            sS[(v + 8 * kh) * 1024 + nCol]        = acc0[v];
            sS[(16 + v + 8 * kh) * 1024 + nCol]   = acc1[v];
        }
    }
    __syncthreads();

    // ---------------- softmax over 1024 cols, 4 rows per wave ----------------
#pragma unroll
    for (int i = 0; i < 4; ++i) {
        const int r = wv * 4 + i;
        float* row = sS + r * 1024;
        float mx = -3.402823466e38f;
#pragma unroll 8
        for (int j = 0; j < 32; ++j) mx = fmaxf(mx, row[lane + 32 * j]);
#pragma unroll
        for (int off = 16; off > 0; off >>= 1) mx = fmaxf(mx, __shfl_xor(mx, off, 32));

        float sum = 0.f;
#pragma unroll 8
        for (int j = 0; j < 32; ++j) {
            float e = __expf(row[lane + 32 * j] - mx);
            row[lane + 32 * j] = e;
            sum += e;
        }
#pragma unroll
        for (int off = 16; off > 0; off >>= 1) sum += __shfl_xor(sum, off, 32);
        const float inv = __frcp_rn(sum);
#pragma unroll 8
        for (int j = 0; j < 32; ++j)
            sB[r * 1024 + lane + 32 * j] = (__bf16)(row[lane + 32 * j] * inv);
    }
    __syncthreads();

    // ---------------- stage 3: o = beta . h_fl  (K = 1024) ----------------
    v8f acc[2][6];
#pragma unroll
    for (int mt = 0; mt < 2; ++mt)
#pragma unroll
        for (int c = 0; c < 6; ++c) acc[mt][c] = zero8();

    const __bf16* hb = h_t + (size_t)b * 768 * 1024;
    for (int kt = 0; kt < 32; ++kt) {
        if (kt + 1 < 32)   // gfx1250 global_prefetch_b8 on next K slab of h
            __builtin_prefetch((const void*)(hb + (size_t)(wv * 6) * 16 * 1024 + (kt + 1) * 32), 0, 3);
        v16bf aB[2];
#pragma unroll
        for (int mt = 0; mt < 2; ++mt)
            aB[mt] = load_a_from_bf16(sB + (mt * 16) * 1024 + kt * 32, 1024, lane);
#pragma unroll
        for (int c = 0; c < 6; ++c) {
            const int colBase = (wv * 6 + c) * 16;   // c' channel tile 0..47
            v16bf bH = load_b_nk(hb + (size_t)colBase * 1024 + kt * 32, 1024, lane);
            acc[0][c] = wmma_bf16(aB[0], bH, acc[0][c]);
            acc[1][c] = wmma_bf16(aB[1], bH, acc[1][c]);
        }
    }

    // ---------------- epilogue: recover layout, gamma*o + x ----------------
    const float gam = gammaP[0];
    const int kh = lane >> 4;
#pragma unroll
    for (int mt = 0; mt < 2; ++mt) {
#pragma unroll
        for (int c = 0; c < 6; ++c) {
            const int cprime = (wv * 6 + c) * 16 + (lane & 15);  // 0..767
            const int sp = cprime >> 8;
            const int ch = cprime & 255;
#pragma unroll
            for (int v = 0; v < 8; ++v) {
                const int n  = nBase + mt * 16 + v + 8 * kh;     // 0..1023
                const int hh = n >> 8;
                const int w  = n & 255;
                const size_t idx =
                    ((size_t)((b * 12 + sp * 4 + hh) * 256 + w)) * 256 + ch;
                out[idx] = gam * acc[mt][c][v] + x[idx];
            }
        }
    }
}

// ---------------------------------------------------------------------------
extern "C" void kernel_launch(void* const* d_in, const int* in_sizes, int n_in,
                              void* d_out, int out_size, void* d_ws, size_t ws_size,
                              hipStream_t stream) {
    (void)in_sizes; (void)n_in; (void)out_size; (void)ws_size;
    const float* x     = (const float*)d_in[0];
    const float* Wf    = (const float*)d_in[1];
    const float* Wg    = (const float*)d_in[2];
    const float* Wh    = (const float*)d_in[3];
    const float* gamma = (const float*)d_in[4];
    float* out = (float*)d_out;

    // workspace carve-up (bf16): f_fl 6.29MB | g_fl 6.29MB | h_t 50.3MB
    __bf16* f_fl = (__bf16*)d_ws;
    __bf16* g_fl = f_fl + (size_t)32 * 1024 * 96;
    __bf16* h_t  = g_fl + (size_t)32 * 1024 * 96;

    salw_proj_kernel<<<dim3(768), dim3(128), 0, stream>>>(x, Wf, Wg, Wh, f_fl, g_fl, h_t);

    const size_t lds = 32 * 1024 * 4 + 32 * 1024 * 2;  // 192 KB (CDNA5 WGP LDS)
    salw_attn_kernel<<<dim3(1024), dim3(256), lds, stream>>>(x, f_fl, g_fl, h_t, gamma, out);
}
